// Attention_50182397886533
// MI455X (gfx1250) — compile-verified, matrix operand order checked by reading, not run on hardware
//
#include <hip/hip_runtime.h>
#include <hip/hip_bf16.h>
#include <stdint.h>

#define BATCH 64
#define NTOK 65
#define DIM 128
#define HEADS 8
#define DHEAD 32
#define INNER 256
#define EPSV 1e-5f
#define MCHUNK 5          // m-values per workgroup
#define NCHUNKS 13        // 13*5 = 65

typedef __attribute__((ext_vector_type(16))) __bf16 v16bf;
typedef __attribute__((ext_vector_type(8)))  __bf16 v8bf;
typedef __attribute__((ext_vector_type(8)))  float  v8f;
typedef __attribute__((ext_vector_type(4)))  float  v4f;
typedef __attribute__((ext_vector_type(4)))  int    v4i;

// ---- workspace layout (bytes) ----
#define WS_XN    0                                  // xn f32 [B,N,128]   = 2,129,920
#define WS_XNT   2129920                            // xnT bf16 [N,B,128] = 1,064,960
#define WS_ATTN  3194880                            // attn f32 [N,N,B,H] = 8,652,800
#define WS_OIN   11847680                           // out_inner f32 [B,N,256] = 4,259,840

__device__ __forceinline__ uint32_t pack_bf16(float a, float b) {
  uint32_t ua = __builtin_bit_cast(uint32_t, a);
  uint32_t ub = __builtin_bit_cast(uint32_t, b);
  return (ua >> 16) | (ub & 0xFFFF0000u);
}

__device__ __forceinline__ v16bf cat8(v8bf lo, v8bf hi) {
  return __builtin_shufflevector(lo, hi, 0,1,2,3,4,5,6,7,8,9,10,11,12,13,14,15);
}

__device__ __forceinline__ v8f v8f_zero() {
  v8f z;
#pragma unroll
  for (int i = 0; i < 8; ++i) z[i] = 0.0f;
  return z;
}

// 16B global -> LDS copy; prefer CDNA5 async-to-LDS path when available.
__device__ __forceinline__ void stage16(const void* g, void* l) {
#if __has_builtin(__builtin_amdgcn_global_load_async_to_lds_b128)
  __builtin_amdgcn_global_load_async_to_lds_b128(
      (__attribute__((address_space(1))) v4i*)g,
      (__attribute__((address_space(3))) v4i*)l, 0, 0);
#else
  *(uint4*)l = *(const uint4*)g;
#endif
}

__device__ __forceinline__ void wait_async() {
#if __has_builtin(__builtin_amdgcn_s_wait_asynccnt)
  __builtin_amdgcn_s_wait_asynccnt(0);
#else
  asm volatile("s_wait_asynccnt 0x0" ::: "memory");
#endif
}

// ---------------- kernel 0: zero the accumulator ----------------
__global__ void pv_zero_kernel(float4* p) {
  p[blockIdx.x * 256 + threadIdx.x] = make_float4(0.f, 0.f, 0.f, 0.f);
}

// ---------------- kernel 1: LayerNorm, write xn f32 + xnT bf16 ----------------
__global__ void pv_ln_kernel(const float* __restrict__ x,
                             const float* __restrict__ g,
                             const float* __restrict__ be,
                             float* __restrict__ xn,
                             unsigned short* __restrict__ xnT) {
  int row = blockIdx.x;            // b*N + n
  int b = row / NTOK, n = row % NTOK;
  int d = threadIdx.x;             // 0..127
  __shared__ float red[DIM];
  float v = x[(size_t)row * DIM + d];
  red[d] = v;
  __syncthreads();
  for (int s = 64; s > 0; s >>= 1) { if (d < s) red[d] += red[d + s]; __syncthreads(); }
  float mu = red[0] * (1.0f / DIM);
  __syncthreads();
  float c = v - mu;
  red[d] = c * c;
  __syncthreads();
  for (int s = 64; s > 0; s >>= 1) { if (d < s) red[d] += red[d + s]; __syncthreads(); }
  float var = red[0] * (1.0f / DIM);
  float y = c * rsqrtf(var + EPSV) * g[d] + be[d];
  xn[(size_t)row * DIM + d] = y;
  // bf16 (round-to-nearest-ish)
  uint32_t bits = __builtin_bit_cast(uint32_t, y);
  bits += 0x7FFFu + ((bits >> 16) & 1u);
  xnT[((size_t)n * BATCH + b) * DIM + d] = (unsigned short)(bits >> 16);
}

// ---------------- kernel 2: QK projection + softmax, write attn[n][m][b][h] ----------------
__global__ void pv_attn_kernel(const float* __restrict__ xn,
                               const float* __restrict__ Wqk,
                               float* __restrict__ attn) {
  int b = blockIdx.x >> 3;
  int h = blockIdx.x & 7;
  int tid = threadIdx.x;           // 256 threads
  __shared__ float qs[NTOK * DHEAD];
  __shared__ float ks[NTOK * DHEAD];
  __shared__ float dots[NTOK * NTOK];
  const float scale = 0.17677669529663687f;   // 32^-0.5
  for (int idx = tid; idx < NTOK * DHEAD; idx += 256) {
    int nr = idx >> 5, c = idx & 31;
    const float* xr = xn + ((size_t)b * NTOK + nr) * DIM;
    int qc = h * DHEAD + c;
    int kc = INNER + h * DHEAD + c;
    float aq = 0.f, ak = 0.f;
    for (int d = 0; d < DIM; ++d) {
      float xv = xr[d];
      aq += xv * Wqk[d * (2 * INNER) + qc];
      ak += xv * Wqk[d * (2 * INNER) + kc];
    }
    qs[idx] = aq * scale;
    ks[idx] = ak;
  }
  __syncthreads();
  for (int idx = tid; idx < NTOK * NTOK; idx += 256) {
    int nq = idx / NTOK, nk = idx % NTOK;
    float s = 0.f;
    for (int c = 0; c < DHEAD; ++c) s += qs[nq * DHEAD + c] * ks[nk * DHEAD + c];
    dots[idx] = s;
  }
  __syncthreads();
  if (tid < NTOK) {
    float mx = -1e30f;
    for (int m = 0; m < NTOK; ++m) mx = fmaxf(mx, dots[tid * NTOK + m]);
    float sum = 0.f;
    for (int m = 0; m < NTOK; ++m) {
      float e = __expf(dots[tid * NTOK + m] - mx);
      dots[tid * NTOK + m] = e;
      sum += e;
    }
    float inv = 1.0f / sum;
    for (int m = 0; m < NTOK; ++m)
      attn[(((size_t)tid * NTOK + m) * BATCH + b) * HEADS + h] = dots[tid * NTOK + m] * inv;
  }
}

// ---------------- kernel 3: fused V-GEMM + attention weighting (the 554 MB stream) ----------------
// grid = (13 m-chunks, 65 n).  256 threads = 8 wave32 waves.
// GEMM per (n,m): A = xn[:,n,:] (64x128 bf16), B = Wv[n,m] (128x256 f32 -> bf16), C = V (64x256 f32)
// then out_acc[b,e] += attn[n][m][b][e>>5] * V[b,e]; atomicAdd into out_inner at the end.
__global__ __launch_bounds__(256) void pv_gemm_kernel(
    const unsigned short* __restrict__ xnT,
    const float* __restrict__ Wv,
    const float* __restrict__ attn,
    float* __restrict__ oin) {
  // B fragments: 64 chunks of 1KB laid out as [ktile(4)][ntile(16)]{h*512 + lane*16}
  __shared__ __align__(16) char Blds[65536];
  // A fragments: [mtile(4)][ktile(4)]{h*512 + lane*16}
  __shared__ __align__(16) char Alds[16384];
  __shared__ __align__(16) float attn_lds[BATCH * HEADS];

  int n = blockIdx.y;
  int chunk = blockIdx.x;
  int tid = threadIdx.x;
  int lane = tid & 31;
  int wave = tid >> 5;

  // ---- stage A tile once (copy-only: async-to-LDS), in WMMA 16-bit A layout ----
#pragma unroll
  for (int i = 0; i < 4; ++i) {
    int s = tid + 256 * i;               // 1024 16B slots
    int ls = s & 31;
    int rest = s >> 5;
    int hh = rest & 1;
    int t = (rest >> 1) & 3;
    int mt = rest >> 3;
    int bb = mt * 16 + (ls & 15);
    int dd = t * 32 + hh * 16 + ((ls >= 16) ? 8 : 0);
    const unsigned short* g = xnT + ((size_t)n * BATCH + bb) * DIM + dd;
    stage16(g, Alds + ((mt * 4 + t) << 10) + (hh << 9) + (ls << 4));
  }

  int mt = wave >> 1;                     // batch tile (4 x 16 rows)
  int uh = wave & 1;                      // which half of the 16 n-tiles

  // base of this wave's attention weights: rows rbase..rbase+7, heads uh*4..uh*4+3
  int rbase = mt * 16 + ((lane >= 16) ? 8 : 0);
  const v4f* wbase = (const v4f*)(attn_lds + rbase * HEADS + uh * 4);

  v8f oacc[8];
#pragma unroll
  for (int i = 0; i < 8; ++i) oacc[i] = v8f_zero();

  for (int mi = 0; mi < MCHUNK; ++mi) {
    int m = chunk * MCHUNK + mi;

    // attn weights for this (n,m): 64x8 f32 = 2KB, async copy
    if (tid < 128) {
      const float* g = attn + ((size_t)n * NTOK + m) * (BATCH * HEADS) + tid * 4;
      stage16(g, (char*)attn_lds + (tid << 4));
    }

    // ---- stage + convert Wv[n][m] (128x256 f32) into bf16 WMMA B-fragment layout ----
    const float* wvbase = Wv + ((size_t)n * NTOK + m) * (DIM * INNER);
    for (int unit = wave; unit < 128; unit += 8) {
      int kg = unit >> 3;                 // k0 = kg*8
      int ng = unit & 7;                  // n0 = ng*32
      int e = ng * 32 + lane;             // coalesced across lanes
      const float* src = wvbase + (size_t)kg * 8 * INNER + e;
      float f0 = src[0 * INNER], f1 = src[1 * INNER], f2 = src[2 * INNER], f3 = src[3 * INNER];
      float f4 = src[4 * INNER], f5 = src[5 * INNER], f6 = src[6 * INNER], f7 = src[7 * INNER];
      uint32_t p0 = pack_bf16(f0, f1), p1 = pack_bf16(f2, f3);
      uint32_t p2 = pack_bf16(f4, f5), p3 = pack_bf16(f6, f7);
      int k0 = kg * 8;
      int t = k0 >> 5;
      int hh = (k0 >> 4) & 1;
      int lhi = (k0 >> 3) & 1;
      int u = e >> 4, nn = e & 15;
      int dl = nn + (lhi << 4);
      uint32_t* dst = (uint32_t*)(Blds + ((t * 16 + u) << 10) + (hh << 9) + (dl << 4));
      dst[0] = p0; dst[1] = p1; dst[2] = p2; dst[3] = p3;   // ds_store_b128
    }
    wait_async();
    __syncthreads();

    // ---- 64x256x128 GEMM: 32 wmma per wave ----
    v8f acc[8];
#pragma unroll
    for (int i = 0; i < 8; ++i) acc[i] = v8f_zero();
#pragma unroll
    for (int t = 0; t < 4; ++t) {
      const char* abase = Alds + ((mt * 4 + t) << 10) + (lane << 4);
      v8bf alo = *(const v8bf*)abase;
      v8bf ahi = *(const v8bf*)(abase + 512);
      v16bf af = cat8(alo, ahi);
#pragma unroll
      for (int nt = 0; nt < 8; ++nt) {
        int u = uh * 8 + nt;
        const char* bbase = Blds + ((t * 16 + u) << 10) + (lane << 4);
        v8bf blo = *(const v8bf*)bbase;
        v8bf bhi = *(const v8bf*)(bbase + 512);
        v16bf bf = cat8(blo, bhi);
        acc[nt] = __builtin_amdgcn_wmma_f32_16x16x32_bf16(
            false, af, false, bf, (short)0, acc[nt], false, false);
      }
    }

    // ---- fold in softmax weight ----
    // row r needs heads uh*4..uh*4+3: one ds_load_b128 at immediate offset r*32B.
#pragma unroll
    for (int r = 0; r < 8; ++r) {
      v4f w4 = wbase[r * 2];              // attn_lds[(rbase+r)*8 + uh*4 .. +3]
#pragma unroll
      for (int nt = 0; nt < 8; ++nt) {
        oacc[nt][r] += w4[nt >> 1] * acc[nt][r];   // head = u>>1, u = uh*8+nt
      }
    }
    __syncthreads();
  }

  // ---- combine partial m-chunks across workgroups ----
  int nn_col = lane & 15;
#pragma unroll
  for (int nt = 0; nt < 8; ++nt) {
    int e = (uh * 8 + nt) * 16 + nn_col;
#pragma unroll
    for (int r = 0; r < 8; ++r) {
      atomicAdd(&oin[((size_t)(rbase + r) * NTOK + n) * INNER + e], oacc[nt][r]);
    }
  }
}

// ---------------- kernel 4: output projection out_inner @ Wout + bout ----------------
__global__ void pv_proj_kernel(const float* __restrict__ oin,
                               const float* __restrict__ Wout,
                               const float* __restrict__ bout,
                               float* __restrict__ out) {
  int row = blockIdx.x;            // b*N + n
  int d = threadIdx.x;             // 0..127
  __shared__ float rbuf[INNER];
  rbuf[d] = oin[(size_t)row * INNER + d];
  rbuf[d + 128] = oin[(size_t)row * INNER + d + 128];
  __syncthreads();
  float acc = bout[d];
  for (int e = 0; e < INNER; ++e) acc += rbuf[e] * Wout[e * DIM + d];
  out[(size_t)row * DIM + d] = acc;
}

extern "C" void kernel_launch(void* const* d_in, const int* in_sizes, int n_in,
                              void* d_out, int out_size, void* d_ws, size_t ws_size,
                              hipStream_t stream) {
  const float* x     = (const float*)d_in[0];
  const float* gamma = (const float*)d_in[1];
  const float* beta  = (const float*)d_in[2];
  const float* Wqk   = (const float*)d_in[3];
  const float* Wv    = (const float*)d_in[4];
  const float* Wout  = (const float*)d_in[5];
  const float* bout  = (const float*)d_in[6];
  float* out = (float*)d_out;
  char* ws = (char*)d_ws;

  float*          xn   = (float*)(ws + WS_XN);
  unsigned short* xnT  = (unsigned short*)(ws + WS_XNT);
  float*          attn = (float*)(ws + WS_ATTN);
  float*          oin  = (float*)(ws + WS_OIN);

  // out_inner accumulator must be zeroed every launch (atomics accumulate into it)
  pv_zero_kernel<<<1040, 256, 0, stream>>>((float4*)oin);          // 1,064,960 floats
  pv_ln_kernel<<<BATCH * NTOK, DIM, 0, stream>>>(x, gamma, beta, xn, xnT);
  pv_attn_kernel<<<BATCH * HEADS, 256, 0, stream>>>(xn, Wqk, attn);
  pv_gemm_kernel<<<dim3(NCHUNKS, NTOK), 256, 0, stream>>>(xnT, Wv, attn, oin);
  pv_proj_kernel<<<BATCH * NTOK, DIM, 0, stream>>>(oin, Wout, bout, out);
}
